// DualLevelAttentionProcessor_76579266887837
// MI455X (gfx1250) — compile-verified, matrix operand order checked by reading, not run on hardware
//
#include <hip/hip_runtime.h>
#include <hip/hip_bf16.h>

// ---------------------------------------------------------------------------
// DualLevelAttentionProcessor for MI455X (gfx1250, wave32, WMMA).
// All matmuls run through v_wmma_f32_16x16x32_f16 (f16 in, f32 accumulate).
// ---------------------------------------------------------------------------

typedef __attribute__((ext_vector_type(16))) _Float16 v16h;
typedef __attribute__((ext_vector_type(8)))  float    v8f;

#define BB   2
#define SS   2048
#define DD   768
#define HH   12
#define NN   128
#define DH   64
#define QLD  2304   // 3*D row stride of QKV buffers

// K-index map for the 16-bit 16x32 A fragment (ISA 7.12.2):
//   VGPR v holds halves {2v, 2v+1}; v in 0..3 -> K base 2v (+8 for lanes 16-31),
//   v in 4..7 -> K base 2v+8 (+8 for lanes 16-31).
// B fragment assumed symmetric (lane%16 -> N, same K map).
__device__ __forceinline__ int kmap(int e, int lane) {
  int v = e >> 1, pos = e & 1, half = lane >> 4;
  return (v << 1) + ((v >= 4) ? 8 : 0) + pos + (half << 3);
}

__device__ __forceinline__ v8f wmma_f16(v16h a, v16h b, v8f c) {
  return __builtin_amdgcn_wmma_f32_16x16x32_f16(false, a, false, b, (short)0, c,
                                                false, false);
}

// ---------------------------------------------------------------------------
// Generic GEMM: C[M,N] = A[M,K] @ W[N,K]^T + bias[N] (+ R[M,N] if R != null)
// block = 256 threads (8 waves). Block tile 64x64, wave tile 16x32 (2 WMMA).
// M % 64 == 0, N % 64 == 0, K % 32 == 0 (true for every call site).
// ---------------------------------------------------------------------------
__global__ __launch_bounds__(256) void k_gemm(
    const float* __restrict__ A, const float* __restrict__ W,
    const float* __restrict__ bias, const float* __restrict__ R,
    float* __restrict__ C, int M, int N, int K) {
  __shared__ _Float16 sA[64][34];
  __shared__ _Float16 sW[64][34];

  const int tid  = threadIdx.x;
  const int lane = tid & 31;
  const int wv   = tid >> 5;       // 0..7
  const int tm   = wv >> 1;        // m-tile 0..3
  const int tn0  = (wv & 1) * 2;   // n-tiles tn0, tn0+1

  const int m0 = blockIdx.y * 64;
  const int n0 = blockIdx.x * 64;

  const int lrow = tid >> 2;       // 0..63
  const int lcol = (tid & 3) * 8;  // 0,8,16,24

  v8f acc0 = {}; v8f acc1 = {};

  for (int k0 = 0; k0 < K; k0 += 32) {
    const float4* ap = (const float4*)(A + (size_t)(m0 + lrow) * K + k0 + lcol);
    const float4* wp = (const float4*)(W + (size_t)(n0 + lrow) * K + k0 + lcol);
    float4 a0 = ap[0], a1 = ap[1];
    float4 w0 = wp[0], w1 = wp[1];
    if (k0 + 32 < K) {  // gfx1250 global_prefetch_b8 of the next K chunk
      __builtin_prefetch((const void*)(A + (size_t)(m0 + lrow) * K + k0 + 32 + lcol), 0, 1);
      __builtin_prefetch((const void*)(W + (size_t)(n0 + lrow) * K + k0 + 32 + lcol), 0, 1);
    }
    __syncthreads();  // previous iteration's LDS reads complete
    sA[lrow][lcol + 0] = (_Float16)a0.x; sA[lrow][lcol + 1] = (_Float16)a0.y;
    sA[lrow][lcol + 2] = (_Float16)a0.z; sA[lrow][lcol + 3] = (_Float16)a0.w;
    sA[lrow][lcol + 4] = (_Float16)a1.x; sA[lrow][lcol + 5] = (_Float16)a1.y;
    sA[lrow][lcol + 6] = (_Float16)a1.z; sA[lrow][lcol + 7] = (_Float16)a1.w;
    sW[lrow][lcol + 0] = (_Float16)w0.x; sW[lrow][lcol + 1] = (_Float16)w0.y;
    sW[lrow][lcol + 2] = (_Float16)w0.z; sW[lrow][lcol + 3] = (_Float16)w0.w;
    sW[lrow][lcol + 4] = (_Float16)w1.x; sW[lrow][lcol + 5] = (_Float16)w1.y;
    sW[lrow][lcol + 6] = (_Float16)w1.z; sW[lrow][lcol + 7] = (_Float16)w1.w;
    __syncthreads();

    const int r  = tm * 16 + (lane & 15);
    const int c0 = tn0 * 16 + (lane & 15);
    const int c1 = (tn0 + 1) * 16 + (lane & 15);
    v16h af, bf0, bf1;
#pragma unroll
    for (int e = 0; e < 16; ++e) {
      int kk = kmap(e, lane);
      af[e]  = sA[r][kk];
      bf0[e] = sW[c0][kk];
      bf1[e] = sW[c1][kk];
    }
    acc0 = wmma_f16(af, bf0, acc0);
    acc1 = wmma_f16(af, bf1, acc1);
  }
  __syncthreads();

  const int rbase = (lane >> 4) * 8;
  const int ccol  = lane & 15;
#pragma unroll
  for (int i = 0; i < 8; ++i) {
    int row = m0 + tm * 16 + rbase + i;
    int c0  = n0 + tn0 * 16 + ccol;
    int c1  = n0 + (tn0 + 1) * 16 + ccol;
    float v0 = acc0[i] + bias[c0];
    float v1 = acc1[i] + bias[c1];
    if (R) {
      v0 += R[(size_t)row * N + c0];
      v1 += R[(size_t)row * N + c1];
    }
    C[(size_t)row * N + c0] = v0;
    C[(size_t)row * N + c1] = v1;
  }
}

// ---------------------------------------------------------------------------
// Fused multi-head attention for one (batch, 16-query tile) block.
// qq  : [B*Lq, 2304]  Q at col h*64,   kv : [B*Lk, 2304]  K at col 768+h*64,
//                                               V at col 1536+h*64.
// outp: [B*Lq, 768]   per-head concat output.
// avgp: [B, Lq, Lk]   head-averaged probabilities (block-exclusive RMW).
// Dynamic LDS: probs[16*Lk] + red[16*17] + outbuf[2*16*64] floats.
// ---------------------------------------------------------------------------
__global__ __launch_bounds__(256) void k_attn(
    const float* __restrict__ qq, const float* __restrict__ kv,
    float* __restrict__ outp, float* __restrict__ avgp,
    int Lq, int Lk, float scale) {
  extern __shared__ float smem[];
  float* probs  = smem;                 // 16 * Lk
  float* red    = probs + 16 * Lk;      // 16 * 17
  float* outbuf = red + 16 * 17;        // 2 * 16 * 64

  const int b    = blockIdx.y;
  const int qt   = blockIdx.x;
  const int tid  = threadIdx.x;
  const int lane = tid & 31;
  const int wv   = tid >> 5;
  const int m16  = lane & 15;
  const int nkt  = Lk >> 4;             // key tiles
  const float invH = 1.0f / (float)HH;

  for (int h = 0; h < HH; ++h) {
    // ---- Q fragments for this head (2 K-chunks of 32), reused across tiles
    v16h qf0, qf1;
    {
      size_t qrow = (size_t)(b * Lq + qt * 16 + m16) * QLD + h * DH;
#pragma unroll
      for (int e = 0; e < 16; ++e) {
        int kk = kmap(e, lane);
        qf0[e] = (_Float16)qq[qrow + kk];
        qf1[e] = (_Float16)qq[qrow + 32 + kk];
      }
    }
    // ---- scores: each wave owns key tiles wv, wv+8, ...
    for (int j = wv; j < nkt; j += 8) {
      size_t krow = (size_t)(b * Lk + j * 16 + m16) * QLD + DD + h * DH;
      v16h bf0, bf1;
#pragma unroll
      for (int e = 0; e < 16; ++e) {
        int kk = kmap(e, lane);
        bf0[e] = (_Float16)kv[krow + kk];
        bf1[e] = (_Float16)kv[krow + 32 + kk];
      }
      v8f c = {};
      c = wmma_f16(qf0, bf0, c);
      c = wmma_f16(qf1, bf1, c);
      const int rbase = (lane >> 4) * 8;
#pragma unroll
      for (int i = 0; i < 8; ++i)
        probs[(rbase + i) * Lk + j * 16 + m16] = c[i] * scale;
    }
    __syncthreads();

    // ---- softmax over each of the 16 rows (16 threads per row)
    const int srow = tid >> 4;
    const int slot = tid & 15;
    float mx = -1e30f;
    for (int k2 = slot; k2 < Lk; k2 += 16)
      mx = fmaxf(mx, probs[srow * Lk + k2]);
    red[srow * 17 + slot] = mx;
    __syncthreads();
    if (slot == 0) {
      float m = -1e30f;
      for (int i = 0; i < 16; ++i) m = fmaxf(m, red[srow * 17 + i]);
      red[srow * 17 + 16] = m;
    }
    __syncthreads();
    const float rowmax = red[srow * 17 + 16];
    float sum = 0.f;
    for (int k2 = slot; k2 < Lk; k2 += 16) {
      float p = __expf(probs[srow * Lk + k2] - rowmax);
      probs[srow * Lk + k2] = p;
      sum += p;
    }
    __syncthreads();
    red[srow * 17 + slot] = sum;
    __syncthreads();
    if (slot == 0) {
      float s = 0.f;
      for (int i = 0; i < 16; ++i) s += red[srow * 17 + i];
      red[srow * 17 + 16] = s;
    }
    __syncthreads();
    const float inv = 1.0f / red[srow * 17 + 16];
    for (int k2 = slot; k2 < Lk; k2 += 16)
      probs[srow * Lk + k2] *= inv;
    __syncthreads();

    // ---- accumulate head-averaged probabilities (block-exclusive region)
    for (int idx = tid; idx < 16 * Lk; idx += 256) {
      int r = idx / Lk, k2 = idx - r * Lk;
      size_t o = (size_t)(b * Lq + qt * 16 + r) * Lk + k2;
      float v = probs[r * Lk + k2] * invH;
      avgp[o] = (h == 0) ? v : (avgp[o] + v);
    }

    // ---- out tile [16 x 64] = probs @ V, split-K over 2 wave groups
    const int nt    = wv & 3;          // dh tile 0..3
    const int halfw = wv >> 2;         // K half 0..1
    const int kbeg  = halfw * (Lk >> 1);
    const int kend  = kbeg + (Lk >> 1);
    v8f c = {};
    for (int kb = kbeg; kb < kend; kb += 32) {
      v16h af, bf;
#pragma unroll
      for (int e = 0; e < 16; ++e) {
        int kk = kmap(e, lane);
        af[e] = (_Float16)probs[m16 * Lk + kb + kk];
        bf[e] = (_Float16)kv[(size_t)(b * Lk + kb + kk) * QLD +
                             2 * DD + h * DH + nt * 16 + m16];
      }
      c = wmma_f16(af, bf, c);
    }
    {
      const int rbase = (lane >> 4) * 8;
#pragma unroll
      for (int i = 0; i < 8; ++i)
        outbuf[halfw * 1024 + (rbase + i) * 64 + nt * 16 + m16] = c[i];
    }
    __syncthreads();
    for (int idx = tid; idx < 1024; idx += 256) {
      int r = idx >> 6, col = idx & 63;
      float v = outbuf[r * 64 + col] + outbuf[1024 + r * 64 + col];
      outp[(size_t)(b * Lq + qt * 16 + r) * DD + h * DH + col] = v;
    }
    __syncthreads();  // probs/outbuf reused next head
  }
}

// ---------------------------------------------------------------------------
// LayerNorm over rows of 768: o = LN(x + r) * g + bb
// ---------------------------------------------------------------------------
__global__ __launch_bounds__(256) void k_ln(
    const float* __restrict__ x, const float* __restrict__ r,
    const float* __restrict__ g, const float* __restrict__ bb,
    float* __restrict__ o) {
  __shared__ float red[256];
  const int tid = threadIdx.x;
  const size_t base = (size_t)blockIdx.x * DD;
  float v0 = x[base + tid]       + r[base + tid];
  float v1 = x[base + 256 + tid] + r[base + 256 + tid];
  float v2 = x[base + 512 + tid] + r[base + 512 + tid];

  red[tid] = v0 + v1 + v2;
  __syncthreads();
  for (int s = 128; s > 0; s >>= 1) {
    if (tid < s) red[tid] += red[tid + s];
    __syncthreads();
  }
  const float mean = red[0] * (1.0f / DD);
  __syncthreads();
  float d0 = v0 - mean, d1 = v1 - mean, d2 = v2 - mean;
  red[tid] = d0 * d0 + d1 * d1 + d2 * d2;
  __syncthreads();
  for (int s = 128; s > 0; s >>= 1) {
    if (tid < s) red[tid] += red[tid + s];
    __syncthreads();
  }
  const float inv = rsqrtf(red[0] * (1.0f / DD) + 1e-5f);
  o[base + tid]       = d0 * inv * g[tid]       + bb[tid];
  o[base + 256 + tid] = d1 * inv * g[256 + tid] + bb[256 + tid];
  o[base + 512 + tid] = d2 * inv * g[512 + tid] + bb[512 + tid];
}

// w[b,k] = mean over q of tok_attn[b,q,k]    (4096 threads)
__global__ __launch_bounds__(256) void k_colmean(
    const float* __restrict__ attn, float* __restrict__ w) {
  int t = blockIdx.x * 256 + threadIdx.x;
  int b = t >> 11, k = t & 2047;
  float s = 0.f;
  for (int q = 0; q < SS; ++q)
    s += attn[((size_t)(b * SS + q)) * SS + k];
  w[t] = s * (1.0f / SS);
}

// node_pre[b,n,:] = sum_s[seg==n] w*x / (sum w + 1e-8). One block per (b,n).
__global__ __launch_bounds__(256) void k_aggregate(
    const float* __restrict__ x, const float* __restrict__ w,
    const int* __restrict__ seg, float* __restrict__ node) {
  const int b = blockIdx.x / NN, n = blockIdx.x % NN;
  const int c = threadIdx.x * 3;
  float a0 = 0.f, a1 = 0.f, a2 = 0.f, den = 0.f;
  for (int s = 0; s < SS; ++s) {
    if (seg[b * SS + s] == n) {
      float wg = w[b * SS + s];
      size_t base = (size_t)(b * SS + s) * DD + c;
      a0 += wg * x[base];
      a1 += wg * x[base + 1];
      a2 += wg * x[base + 2];
      den += wg;
    }
  }
  float inv = 1.0f / (den + 1e-8f);
  size_t ob = (size_t)(b * NN + n) * DD + c;
  node[ob] = a0 * inv; node[ob + 1] = a1 * inv; node[ob + 2] = a2 * inv;
}

// enhanced = tok2 + node_emb[seg]   (flat over B*S*D)
__global__ __launch_bounds__(256) void k_gather(
    const float* __restrict__ tok2, const float* __restrict__ node,
    const int* __restrict__ seg, float* __restrict__ out) {
  size_t t = (size_t)blockIdx.x * 256 + threadIdx.x;
  int c = (int)(t % DD);
  int bs = (int)(t / DD);
  int b = bs / SS;
  int sg = seg[bs];
  out[t] = tok2[t] + node[(size_t)(b * NN + sg) * DD + c];
}

// ---------------------------------------------------------------------------
extern "C" void kernel_launch(void* const* d_in, const int* in_sizes, int n_in,
                              void* d_out, int out_size, void* d_ws, size_t ws_size,
                              hipStream_t stream) {
  (void)in_sizes; (void)n_in; (void)out_size; (void)ws_size;
  const float* hs     = (const float*)d_in[0];
  const int*   seg    = (const int*)d_in[1];
  const float* Wqkv_t = (const float*)d_in[2];
  const float* bqkv_t = (const float*)d_in[3];
  const float* Wo_t   = (const float*)d_in[4];
  const float* bo_t   = (const float*)d_in[5];
  const float* Wqkv_c = (const float*)d_in[6];
  const float* bqkv_c = (const float*)d_in[7];
  const float* Wo_c   = (const float*)d_in[8];
  const float* bo_c   = (const float*)d_in[9];
  const float* g_tok  = (const float*)d_in[10];
  const float* b_tok  = (const float*)d_in[11];
  const float* g_node = (const float*)d_in[12];
  const float* b_node = (const float*)d_in[13];
  const float* W_out  = (const float*)d_in[14];
  const float* b_out  = (const float*)d_in[15];

  // d_out segments (flat fp32 concat, reference return order)
  float* out0      = (float*)d_out;                  // [B,S,D]
  float* node_emb  = out0 + (size_t)BB * SS * DD;    // [B,N,D]
  float* tok_attn  = node_emb + (size_t)BB * NN * DD;// [B,S,S]
  float* node_attn = tok_attn + (size_t)BB * SS * SS;// [B,N,N]
  float* cross_w   = node_attn + (size_t)BB * NN * NN;// [B,S,N]

  // workspace layout (fp32)
  float* ws = (float*)d_ws;
  float* qkv      = ws;                               size_t o = (size_t)BB * SS * QLD;
  float* attnout  = ws + o;                           o += (size_t)BB * SS * DD;
  float* proj     = ws + o;                           o += (size_t)BB * SS * DD;
  float* tokout   = ws + o;                           o += (size_t)BB * SS * DD;
  float* tokout2  = ws + o;                           o += (size_t)BB * SS * DD;
  float* wbuf     = ws + o;                           o += (size_t)BB * SS;
  float* nqkv     = ws + o;                           o += (size_t)BB * NN * QLD;
  float* node_pre = ws + o;                           o += (size_t)BB * NN * DD;
  float* nA       = ws + o;                           o += (size_t)BB * NN * DD;
  float* nB       = ws + o;                           /* o += BB*NN*DD; */

  const dim3 blk(256);
  const float scale = 0.125f;  // 1/sqrt(64)
  const size_t sh_tok  = (size_t)(16 * SS + 16 * 17 + 2048) * sizeof(float);
  const size_t sh_node = (size_t)(16 * NN + 16 * 17 + 2048) * sizeof(float);

  // 1) token QKV = hs @ Wqkv_t^T + bqkv_t
  k_gemm<<<dim3(QLD / 64, (BB * SS) / 64), blk, 0, stream>>>(
      hs, Wqkv_t, bqkv_t, nullptr, qkv, BB * SS, QLD, DD);
  // 2) token self-attention (+ head-averaged tok_attn)
  k_attn<<<dim3(SS / 16, BB), blk, sh_tok, stream>>>(
      qkv, qkv, attnout, tok_attn, SS, SS, scale);
  // 3) Wo projection
  k_gemm<<<dim3(DD / 64, (BB * SS) / 64), blk, 0, stream>>>(
      attnout, Wo_t, bo_t, nullptr, proj, BB * SS, DD, DD);
  // 4) token_output = LN(hs + proj)
  k_ln<<<BB * SS, blk, 0, stream>>>(hs, proj, g_tok, b_tok, tokout);
  // 5) w = column mean of tok_attn
  k_colmean<<<(BB * SS) / 256, blk, 0, stream>>>(tok_attn, wbuf);
  // 6) weighted segment aggregation -> node_pre
  k_aggregate<<<BB * NN, blk, 0, stream>>>(tokout, wbuf, seg, node_pre);
  // 7) node QKV (token weights)
  k_gemm<<<dim3(QLD / 64, (BB * NN) / 64), blk, 0, stream>>>(
      node_pre, Wqkv_t, bqkv_t, nullptr, nqkv, BB * NN, QLD, DD);
  // 8) node self-attention (+ node_attn)
  k_attn<<<dim3(NN / 16, BB), blk, sh_node, stream>>>(
      nqkv, nqkv, nA, node_attn, NN, NN, scale);
  // 9) node Wo projection
  k_gemm<<<dim3(DD / 64, (BB * NN) / 64), blk, 0, stream>>>(
      nA, Wo_t, bo_t, nullptr, nB, BB * NN, DD, DD);
  // 10) node_emb = LN(node_pre + nB)  -> output segment
  k_ln<<<BB * NN, blk, 0, stream>>>(node_pre, nB, g_node, b_node, node_emb);
  // 11) cross QKV on tokens (Q part used)
  k_gemm<<<dim3(QLD / 64, (BB * SS) / 64), blk, 0, stream>>>(
      tokout, Wqkv_c, bqkv_c, nullptr, qkv, BB * SS, QLD, DD);
  // 12) cross QKV on nodes (K,V parts used)
  k_gemm<<<dim3(QLD / 64, (BB * NN) / 64), blk, 0, stream>>>(
      node_emb, Wqkv_c, bqkv_c, nullptr, nqkv, BB * NN, QLD, DD);
  // 13) cross attention tokens->nodes (+ cross_w)
  k_attn<<<dim3(SS / 16, BB), blk, sh_node, stream>>>(
      qkv, nqkv, attnout, cross_w, SS, NN, scale);
  // 14) token_output += cross_out @ Wo_c^T + bo_c (residual in GEMM)
  k_gemm<<<dim3(DD / 64, (BB * SS) / 64), blk, 0, stream>>>(
      attnout, Wo_c, bo_c, tokout, tokout2, BB * SS, DD, DD);
  // 15) enhanced = tokout2 + node_emb[seg]
  k_gather<<<(BB * SS * DD) / 256, blk, 0, stream>>>(tokout2, node_emb, seg, proj);
  // 16) output = enhanced @ W_out^T + b_out
  k_gemm<<<dim3(DD / 64, (BB * SS) / 64), blk, 0, stream>>>(
      proj, W_out, b_out, nullptr, out0, BB * SS, DD, DD);
}
